// ResidualVQ_79353815761108
// MI455X (gfx1250) — matrix operand order repaired and checked by reading, not compiled, not run on hardware
//
#include <hip/hip_runtime.h>

typedef __attribute__((ext_vector_type(16))) __bf16  v16bf;
typedef __attribute__((ext_vector_type(16))) short   v16s;
typedef __attribute__((ext_vector_type(8)))  float   v8f;

#define D_  1024
#define CD_ 16
#define K_  1024
#define L_  8
#define NTOK 32768
#define RS 1028   // residual LDS row stride (floats), padded vs 64 banks
#define SS 1028   // score   LDS row stride (floats)

// ---- WMMA operand builders (CDNA5 16x16x32 bf16 layouts, wave32) ----

// A-matrix 16x32 bf16 from f32 LDS tile (row-major, stride `stride`):
// lanes 0-15: K = k0+0..15 (elem e == K-off e)
// lanes16-31: K = k0+16..31
__device__ __forceinline__ v16bf ldsA_bf16(const float* rowbase, int stride, int k0, int lane){
  const int m  = lane & 15;
  const int hi = lane >> 4;
  const float* p = rowbase + m * stride + k0 + hi * 16;
  v16bf a;
#pragma unroll
  for (int j = 0; j < 8; ++j) a[j]     = (__bf16)p[j];
#pragma unroll
  for (int j = 0; j < 8; ++j) a[8 + j] = (__bf16)p[8 + j];
  return a;
}

// A-matrix with true K=16, zero-padded to 32 (z_i / z_q 16x16 tiles in LDS)
__device__ __forceinline__ v16bf ldsA16_pad(const float (*Z)[20], int lane){
  const int m  = lane & 15;
  const int hi = lane >> 4;
  v16bf a;
#pragma unroll
  for (int j = 0; j < 8; ++j) a[j]     = hi ? (__bf16)0.0f : (__bf16)Z[m][j];
#pragma unroll
  for (int j = 0; j < 8; ++j) a[8 + j] = hi ? (__bf16)0.0f : (__bf16)Z[m][8 + j];
  return a;
}

// B-matrix 32x16 bf16 from pre-packed bf16 global (B[k][n] = W[n][k], ld=ldb elems):
// lanes 0-15 hold K=k0+0..15 (elem e == K-off e), lanes16-31 hold K=k0+16..31
__device__ __forceinline__ v16bf gB_bf16(const unsigned short* base, int ldb, int k0, int lane){
  const int n  = lane & 15;
  const int hi = lane >> 4;
  v16s raw = *(const v16s*)(base + n * ldb + k0 + hi * 16);
  return __builtin_bit_cast(v16bf, raw);
}

// B-matrix true K=16 (pad to 32) from pre-packed bf16 rows (16 elems/row contiguous)
__device__ __forceinline__ v16bf gB16_bf16(const unsigned short* rowblock, int lane){
  v16bf b;
  if (lane < 16) {
    v16s raw = *(const v16s*)(rowblock + (lane & 15) * 16);
    b = __builtin_bit_cast(v16bf, raw);
  } else {
#pragma unroll
    for (int j = 0; j < 16; ++j) b[j] = (__bf16)0.0f;
  }
  return b;
}

__device__ __forceinline__ v8f wmma_bf16(v16bf a, v16bf b, v8f c){
  return __builtin_amdgcn_wmma_f32_16x16x32_bf16(false, a, false, b, (short)0, c, false, false);
}

__device__ __forceinline__ unsigned short bf16_bits(float f){
  const __bf16 h = (__bf16)f;
  return __builtin_bit_cast(unsigned short, h);
}

// ---- prep: bf16-pack Wd/Wu, L2-normalize codebook -> bf16, zero loss outputs ----
extern "C" __global__ void rvq_prep(const float* __restrict__ cb,
                                    const float* __restrict__ Wd,
                                    const float* __restrict__ Wu,
                                    unsigned short* __restrict__ cbn,
                                    unsigned short* __restrict__ Wdb,
                                    unsigned short* __restrict__ Wub,
                                    float* __restrict__ out_scal)
{
  const int t = blockIdx.x * blockDim.x + threadIdx.x;   // 0..131071
  if (t == 0) { out_scal[0] = 0.f; out_scal[1] = 0.f; out_scal[2] = 0.f; }

  if (t < L_ * CD_ * D_) {          // elementwise weight packing (layout unchanged)
    Wdb[t] = bf16_bits(Wd[t]);
    Wub[t] = bf16_bits(Wu[t]);
  }
  if (t < L_ * K_) {                // codebook row L2-normalize -> bf16
    const float* p = cb + t * CD_;
    float ss = 0.f;
#pragma unroll
    for (int c = 0; c < CD_; ++c) ss += p[c] * p[c];
    const float inv = 1.0f / fmaxf(sqrtf(ss), 1e-12f);
#pragma unroll
    for (int c = 0; c < CD_; ++c) cbn[t * CD_ + c] = bf16_bits(p[c] * inv);
  }
}

// ---- main: 16 tokens per workgroup, residual resident in LDS across 8 levels ----
extern "C" __global__ __launch_bounds__(256)
void rvq_main(const float* __restrict__ z_e,
              const float* __restrict__ bd,
              const float* __restrict__ cb,
              const float* __restrict__ bu,
              const unsigned short* __restrict__ cbn,
              const unsigned short* __restrict__ Wdb,
              const unsigned short* __restrict__ Wub,
              float* __restrict__ out_zq,
              float* __restrict__ out_codes,
              float* __restrict__ out_scal)
{
  __shared__ float R[16 * RS];          // residual tile (f32 master copy)
  __shared__ float S[16 * SS];          // scores vs 1024 codes
  __shared__ float zi[16][20];
  __shared__ float zq[16][20];
  __shared__ float bestv[16][16];
  __shared__ int   besti[16][16];
  __shared__ int   codes_s[16];
  __shared__ float loss_s;

  const int tid  = threadIdx.x;
  const int lane = tid & 31;
  const int wave = tid >> 5;                     // 0..7
  const long token0 = (long)blockIdx.x * 16;

  // ---- async load residual tile: global -> LDS directly (ASYNCcnt path) ----
  {
    const int m  = tid >> 4;
    const int c0 = (tid & 15) * 64;              // 64 floats = 256 B per thread
    const float* src = z_e + (token0 + m) * D_ + c0;
    const unsigned lds0 = (unsigned)(uintptr_t)&R[m * RS + c0];
#pragma unroll
    for (int j = 0; j < 64; j += 4) {
      asm volatile("global_load_async_to_lds_b128 %0, %1, off"
                   :: "v"(lds0 + (unsigned)(j * 4)), "v"(src + j) : "memory");
    }
  }
  if (tid == 0) loss_s = 0.0f;
  asm volatile("s_wait_asynccnt 0x0" ::: "memory");
  __syncthreads();

  for (int li = 0; li < L_; ++li) {
    const float*          bdL  = bd  + li * CD_;
    const float*          cbL  = cb  + li * K_ * CD_;
    const float*          buL  = bu  + li * D_;
    const unsigned short* cbnL = cbn + li * K_ * CD_;
    const unsigned short* WdbL = Wdb + li * CD_ * D_;
    const unsigned short* WubL = Wub + li * D_ * CD_;

    // z_i := bd
    zi[tid >> 4][tid & 15] = bdL[tid & 15];
    __syncthreads();

    // ---- proj_down: z_i += residual(16x1024) @ Wd^T(1024x16) ----
    {
      v8f acc = {0.f,0.f,0.f,0.f,0.f,0.f,0.f,0.f};
#pragma unroll
      for (int s = 0; s < 4; ++s) {
        const int k0 = wave * 128 + s * 32;
        v16bf a = ldsA_bf16(R, RS, k0, lane);
        v16bf b = gB_bf16(WdbL, D_, k0, lane);
        acc = wmma_bf16(a, b, acc);
      }
      const int n  = lane & 15;
      const int mb = (lane >> 4) * 8;            // D-layout: VGPR v -> row mb+v
#pragma unroll
      for (int v = 0; v < 8; ++v)
        atomicAdd(&zi[mb + v][n], acc[v]);       // ds_add_f32 cross-wave reduce
    }
    __syncthreads();

    // ---- scores: S = z_i(16x16) @ cbn^T(16x1024)  (argmin d2 == argmax score) ----
    {
      v16bf a = ldsA16_pad(zi, lane);
#pragma unroll
      for (int cc = 0; cc < 8; ++cc) {
        const int chunk = wave * 8 + cc;         // 16 codes per chunk
        v16bf b = gB16_bf16(cbnL + chunk * 16 * CD_, lane);
        v8f d = {0.f,0.f,0.f,0.f,0.f,0.f,0.f,0.f};
        d = wmma_bf16(a, b, d);
        const int col = chunk * 16 + (lane & 15);
        const int mb  = (lane >> 4) * 8;
#pragma unroll
        for (int v = 0; v < 8; ++v) S[(mb + v) * SS + col] = d[v];
      }
    }
    __syncthreads();

    // ---- argmax over 1024 codes per token (16 threads/token x 64 codes, b128 scans) ----
    {
      const int r  = tid >> 4;
      const int j  = tid & 15;
      const int kb = j * 64;
      const float* Sr = &S[r * SS + kb];
      float best = -3.4e38f; int bi = kb;
#pragma unroll 4
      for (int k = 0; k < 64; k += 4) {
        const float4 s4 = *(const float4*)(Sr + k);
        if (s4.x > best) { best = s4.x; bi = kb + k; }
        if (s4.y > best) { best = s4.y; bi = kb + k + 1; }
        if (s4.z > best) { best = s4.z; bi = kb + k + 2; }
        if (s4.w > best) { best = s4.w; bi = kb + k + 3; }
      }
      bestv[r][j] = best; besti[r][j] = bi;
    }
    __syncthreads();
    if ((tid & 15) == 0) {
      const int r = tid >> 4;
      float best = bestv[r][0]; int bi = besti[r][0];
#pragma unroll
      for (int j = 1; j < 16; ++j) {
        const float v = bestv[r][j]; const int ix = besti[r][j];
        if (v > best || (v == best && ix < bi)) { best = v; bi = ix; }
      }
      codes_s[r] = bi;
      out_codes[(token0 + r) * L_ + li] = (float)bi;
    }
    __syncthreads();

    // ---- gather raw codebook row, emit z_q_concat, accumulate loss ----
    {
      const int r = tid >> 4, c = tid & 15;
      const float q = cbL[codes_s[r] * CD_ + c];
      zq[r][c] = q;
      out_zq[(token0 + r) * (L_ * CD_) + li * CD_ + c] = q;  // straight-through fwd value
      const float df = zi[r][c] - q;
      atomicAdd(&loss_s, df * df);
    }
    __syncthreads();

    // ---- residual -= z_q(16x16) @ Wu^T(16x1024) + bu ----
    {
      v16bf a = ldsA16_pad(zq, lane);
#pragma unroll
      for (int cc = 0; cc < 8; ++cc) {
        const int n0 = (wave * 8 + cc) * 16;
        v16bf b = gB16_bf16(WubL + n0 * CD_, lane);
        v8f d = {0.f,0.f,0.f,0.f,0.f,0.f,0.f,0.f};
        d = wmma_bf16(a, b, d);
        const int col = n0 + (lane & 15);
        const float buv = buL[col];
        const int mb = (lane >> 4) * 8;
#pragma unroll
        for (int v = 0; v < 8; ++v) {
          float* rp = &R[(mb + v) * RS + col];
          *rp = *rp - d[v] - buv;
        }
      }
    }
    __syncthreads();
  }

  if (tid == 0) {
    const float sc = 1.0f / ((float)NTOK * (float)CD_);
    atomicAdd(&out_scal[0], loss_s * sc);   // commitment_loss
    atomicAdd(&out_scal[1], loss_s * sc);   // codebook_loss (same fwd value)
  }
}

extern "C" void kernel_launch(void* const* d_in, const int* in_sizes, int n_in,
                              void* d_out, int out_size, void* d_ws, size_t ws_size,
                              hipStream_t stream)
{
  const float* z_e = (const float*)d_in[0];
  const float* Wd  = (const float*)d_in[1];
  const float* bd  = (const float*)d_in[2];
  const float* cb  = (const float*)d_in[3];
  const float* Wu  = (const float*)d_in[4];
  const float* bu  = (const float*)d_in[5];

  float* out       = (float*)d_out;
  float* out_zq    = out;                                 // 8*4096*128
  float* out_codes = out + 8 * 4096 * 128;                // 8*4096*8
  float* out_scal  = out + 8 * 4096 * 128 + 8 * 4096 * 8; // commit, cbl, entropy

  // workspace: bf16 packs (256 KB each)
  unsigned short* cbn = (unsigned short*)d_ws;            // [L][K][CD]  normalized
  unsigned short* Wdb = cbn + L_ * K_ * CD_;              // [L][CD][D]
  unsigned short* Wub = Wdb + L_ * CD_ * D_;              // [L][D][CD]

  rvq_prep<<<512, 256, 0, stream>>>(cb, Wd, Wu, cbn, Wdb, Wub, out_scal);
  rvq_main<<<NTOK / 16, 256, 0, stream>>>(z_e, bd, cb, bu, cbn, Wdb, Wub,
                                          out_zq, out_codes, out_scal);
}
